// VanillaAttention_42614665511009
// MI455X (gfx1250) — matrix operand order repaired and checked by reading, not compiled, not run on hardware
//
#include <hip/hip_runtime.h>
#include <hip/hip_bf16.h>
#include <stdint.h>

// ---------------------------------------------------------------------------
// Types
// ---------------------------------------------------------------------------
typedef unsigned short BF;                                        // raw bf16
typedef __attribute__((ext_vector_type(16))) __bf16        v16bf; // WMMA A/B
typedef __attribute__((ext_vector_type(8)))  float         v8f;   // WMMA C/D
typedef __attribute__((ext_vector_type(4)))  unsigned int  v4u;   // 16B chunk

#define NTOK   1024      // x*y tokens
#define CDIM   512
#define HEADS  8
#define DHEAD  64
#define BATCH  8
#define SCALE  0.125f    // 64^-0.5

// ---------------------------------------------------------------------------
// bf16 helpers (round-to-nearest-even)
// ---------------------------------------------------------------------------
__device__ __forceinline__ BF f2bf(float f) {
  unsigned u = __float_as_uint(f);
  u += 0x7FFFu + ((u >> 16) & 1u);
  return (BF)(u >> 16);
}
__device__ __forceinline__ float bf2f(BF h) {
  return __uint_as_float(((unsigned)h) << 16);
}

// ---------------------------------------------------------------------------
// WMMA fragment loads.  GEMM is "NT": A row-major [M,K], Bt row-major [N,K].
// v_wmma_f32_16x16x32_bf16 wave32 layouts as used here (A/B k order is contiguous, not the ISA table order):
//   A: lane(m=ln, half) elem i    -> K=kb+half*16+i
//   B: lane(n=ln, half) elem i    -> K=kb+half*16+i
//   D: lane(n=ln, half) elem j    -> row half*8+j
// ---------------------------------------------------------------------------
union FragU { v4u q[2]; v16bf v; };

__device__ __forceinline__ v16bf load_a_frag(const BF* base, int row, int ld,
                                             int kb, int half) {
  const BF* p = base + (size_t)row * ld + kb + half * 16;
  FragU u;
  u.q[0] = *(const v4u*)(p);
  u.q[1] = *(const v4u*)(p + 8);
  return u.v;
}
__device__ __forceinline__ v16bf load_b_frag(const BF* base, int row, int ld,
                                             int kb, int half) {
  const BF* p = base + (size_t)row * ld + kb + half * 16;
  FragU u;
  u.q[0] = *(const v4u*)(p);
  u.q[1] = *(const v4u*)(p + 8);
  return u.v;
}
__device__ __forceinline__ v8f wmma_bf16(v16bf a, v16bf b, v8f c) {
  return __builtin_amdgcn_wmma_f32_16x16x32_bf16(false, a, false, b,
                                                 (short)0, c, false, false);
}

// ---------------------------------------------------------------------------
// CDNA5 async global->LDS (guarded; falls back to sync copy).
// Builtin signature (from compiler diagnostic): param0 = AS1 ptr to
// "int __vector(4)" (vector_size style), param1 = AS3 ptr, then 2 imm ints.
// ---------------------------------------------------------------------------
#if __has_builtin(__builtin_amdgcn_global_load_async_to_lds_b128)
#define HAVE_ASYNC_G2L 1
typedef int v4i_vs __attribute__((vector_size(16)));
typedef __attribute__((address_space(1))) v4i_vs as1_v4i;
typedef __attribute__((address_space(3))) v4i_vs as3_v4i;
#endif

__device__ __forceinline__ void g2l_16B(BF* lds, const BF* g) {
#if defined(HAVE_ASYNC_G2L)
  __builtin_amdgcn_global_load_async_to_lds_b128((as1_v4i*)g, (as3_v4i*)lds,
                                                 0, 0);
#else
  *(v4u*)lds = *(const v4u*)g;
#endif
}
__device__ __forceinline__ void g2l_wait() {
#if defined(HAVE_ASYNC_G2L)
#if __has_builtin(__builtin_amdgcn_s_wait_asynccnt)
  __builtin_amdgcn_s_wait_asynccnt(0);
#else
  asm volatile("s_wait_asynccnt 0" ::: "memory");
#endif
#endif
}

// ---------------------------------------------------------------------------
// Conversion kernels
// ---------------------------------------------------------------------------
__global__ void cvt_kernel(const float* __restrict__ in, BF* __restrict__ out,
                           int n) {
  int i = blockIdx.x * 256 + threadIdx.x;
  if (i < n) out[i] = f2bf(in[i]);
}

// fmap [b][c][n] f32  ->  fT [b][n][c] bf16   (B operand of QKVL projection)
__global__ void transpose_fmap_kernel(const float* __restrict__ fmap,
                                      BF* __restrict__ fT) {
  int i = blockIdx.x * 256 + threadIdx.x;
  if (i >= BATCH * NTOK * CDIM) return;
  int c = i & (CDIM - 1);
  int t = i >> 9;
  int n = t & (NTOK - 1);
  int b = t >> 10;
  fT[i] = f2bf(fmap[((size_t)b * CDIM + c) * NTOK + n]);
}

// ---------------------------------------------------------------------------
// Kernel A: QKVL projection.  out[b][o][n] = sum_c W[o][c] * fmapT[b][n][c]
// One wave computes a 64x64 tile.  Scatter-stores into attention layouts:
//   q  -> [bh][n][d] bf16 (pre-scaled by SCALE)
//   k  -> [bh][n][d] bf16
//   vT -> [bh][d][n] bf16
//   L  -> Y[b][n][c] bf16  (attention kernel accumulates on top)
// ---------------------------------------------------------------------------
__global__ __launch_bounds__(32) void
qkvl_kernel(const BF* __restrict__ Wbf, const BF* __restrict__ fT,
            BF* __restrict__ qb, BF* __restrict__ kb, BF* __restrict__ vT,
            BF* __restrict__ Y) {
  const int lane = threadIdx.x & 31, half = lane >> 4, ln = lane & 15;
  const int n0 = blockIdx.x * 64;
  const int m0 = blockIdx.y * 64;
  const int b  = blockIdx.z;
  const BF* Bt = fT + (size_t)b * NTOK * CDIM;

  v8f acc[4][4] = {};
  for (int k0 = 0; k0 < CDIM; k0 += 32) {
    v16bf af[4], bfr[4];
#pragma unroll
    for (int mi = 0; mi < 4; ++mi)
      af[mi] = load_a_frag(Wbf, m0 + mi * 16 + ln, CDIM, k0, half);
#pragma unroll
    for (int ni = 0; ni < 4; ++ni)
      bfr[ni] = load_b_frag(Bt, n0 + ni * 16 + ln, CDIM, k0, half);
#pragma unroll
    for (int mi = 0; mi < 4; ++mi)
#pragma unroll
      for (int ni = 0; ni < 4; ++ni)
        acc[mi][ni] = wmma_bf16(af[mi], bfr[ni], acc[mi][ni]);
  }

#pragma unroll
  for (int mi = 0; mi < 4; ++mi)
#pragma unroll
    for (int ni = 0; ni < 4; ++ni)
#pragma unroll
      for (int j = 0; j < 8; ++j) {
        int   m = m0 + mi * 16 + half * 8 + j;
        int   n = n0 + ni * 16 + ln;
        float v = acc[mi][ni][j];
        int chunk = m >> 9, r = m & 511, h = r >> 6, d = r & 63;
        size_t bh = (size_t)(b * HEADS + h);
        if (chunk == 0)
          qb[(bh * NTOK + n) * DHEAD + d] = f2bf(v * SCALE);
        else if (chunk == 1)
          kb[(bh * NTOK + n) * DHEAD + d] = f2bf(v);
        else if (chunk == 2)
          vT[(bh * DHEAD + d) * NTOK + n] = f2bf(v);
        else
          Y[((size_t)b * NTOK + n) * CDIM + r] = f2bf(v);
      }
}

// ---------------------------------------------------------------------------
// Kernel B: flash-style attention.  WG = (bh, 64-query block), 4 waves.
// Each wave owns 16 query rows; iterates 64-key blocks with online softmax.
// Y[b][n][h*64+d] += attention output (Y already holds L).
// ---------------------------------------------------------------------------
__global__ __launch_bounds__(128) void
attn_kernel(const BF* __restrict__ qb, const BF* __restrict__ kb,
            const BF* __restrict__ vT, BF* __restrict__ Y) {
  const int tid  = threadIdx.x;
  const int wid  = tid >> 5, lane = tid & 31;
  const int half = lane >> 4, ln = lane & 15;
  const int bh   = blockIdx.x;           // 0..63
  const int qblk = blockIdx.y;           // 0..15
  const int b = bh >> 3, h = bh & 7;

  __shared__ __align__(16) BF qs[64 * 64];       // staged Q block (8 KB)
  __shared__ __align__(16) BF ps[4][16 * 64];    // per-wave P tile (bf16)

  // --- stage Q block into LDS with async copies (512 x 16B chunks) ---------
  const BF* qsrc = qb + ((size_t)bh * NTOK + (size_t)qblk * 64) * DHEAD;
#pragma unroll
  for (int it = 0; it < 4; ++it) {
    int chunk = it * 128 + tid;
    g2l_16B(&qs[chunk * 8], qsrc + chunk * 8);
  }
  g2l_wait();
  __syncthreads();

  v16bf qa[2];
#pragma unroll
  for (int ks = 0; ks < 2; ++ks)
    qa[ks] = load_a_frag(qs, wid * 16 + ln, DHEAD, ks * 32, half);

  v8f o[4] = {};
  float rmax[8], rsum[8];
#pragma unroll
  for (int j = 0; j < 8; ++j) { rmax[j] = -1e30f; rsum[j] = 0.f; }

  const BF* Kb = kb + (size_t)bh * NTOK * DHEAD;   // [n][d]
  const BF* Vb = vT + (size_t)bh * DHEAD * NTOK;   // [d][n]

  for (int kb0 = 0; kb0 < NTOK; kb0 += 64) {
    // ---- S tile (16 x 64): q  @ k^T --------------------------------------
    v8f s[4] = {};
#pragma unroll
    for (int ni = 0; ni < 4; ++ni)
#pragma unroll
      for (int ks = 0; ks < 2; ++ks) {
        v16bf bk = load_b_frag(Kb, kb0 + ni * 16 + ln, DHEAD, ks * 32, half);
        s[ni] = wmma_bf16(qa[ks], bk, s[ni]);
      }

    // ---- block row-max (across 4 frags, then 16 lanes of half-group) -----
    float nm[8];
#pragma unroll
    for (int j = 0; j < 8; ++j) {
      nm[j] = fmaxf(fmaxf(s[0][j], s[1][j]), fmaxf(s[2][j], s[3][j]));
#pragma unroll
      for (int d = 1; d < 16; d <<= 1)
        nm[j] = fmaxf(nm[j], __shfl_xor(nm[j], d, 32));
    }
    float cf[8];
#pragma unroll
    for (int j = 0; j < 8; ++j) {
      float nx = fmaxf(rmax[j], nm[j]);
      cf[j] = __expf(rmax[j] - nx);
      rmax[j] = nx;
    }

    // ---- P = exp(S - max), write bf16 tile to LDS, row sums --------------
    float psum[8];
#pragma unroll
    for (int j = 0; j < 8; ++j) psum[j] = 0.f;
#pragma unroll
    for (int ni = 0; ni < 4; ++ni)
#pragma unroll
      for (int j = 0; j < 8; ++j) {
        float p = __expf(s[ni][j] - rmax[j]);
        psum[j] += p;
        ps[wid][(half * 8 + j) * 64 + ni * 16 + ln] = f2bf(p);
      }
#pragma unroll
    for (int j = 0; j < 8; ++j) {
#pragma unroll
      for (int d = 1; d < 16; d <<= 1) psum[j] += __shfl_xor(psum[j], d, 32);
      rsum[j] = rsum[j] * cf[j] + psum[j];
    }
#pragma unroll
    for (int ni = 0; ni < 4; ++ni)
#pragma unroll
      for (int j = 0; j < 8; ++j) o[ni][j] *= cf[j];

    // ---- O += P @ V  (A frags from LDS, B frags from vT) -----------------
#pragma unroll
    for (int ks = 0; ks < 2; ++ks) {
      v16bf pa = load_a_frag(ps[wid], ln, 64, ks * 32, half);
#pragma unroll
      for (int ni = 0; ni < 4; ++ni) {
        v16bf vb = load_b_frag(Vb, ni * 16 + ln, NTOK, kb0 + ks * 32, half);
        o[ni] = wmma_bf16(pa, vb, o[ni]);
      }
    }
  }

  // ---- epilogue: normalize, add L (already in Y), store bf16 -------------
  const size_t ybase =
      ((size_t)b * NTOK + (size_t)qblk * 64 + wid * 16) * CDIM + h * DHEAD;
#pragma unroll
  for (int ni = 0; ni < 4; ++ni)
#pragma unroll
    for (int j = 0; j < 8; ++j) {
      int    row = half * 8 + j;
      int    col = ni * 16 + ln;
      size_t idx = ybase + (size_t)row * CDIM + col;
      float  Lv  = bf2f(Y[idx]);
      Y[idx] = f2bf(o[ni][j] / rsum[j] + Lv);
    }
}

// ---------------------------------------------------------------------------
// Kernel C: output projection.  out[b][o][n] = sum_c Wout[o][c]*Y[b][n][c]+bout
// ---------------------------------------------------------------------------
__global__ __launch_bounds__(32) void
outproj_kernel(const BF* __restrict__ Wbf, const BF* __restrict__ Y,
               const float* __restrict__ bout, float* __restrict__ out) {
  const int lane = threadIdx.x & 31, half = lane >> 4, ln = lane & 15;
  const int n0 = blockIdx.x * 64;
  const int m0 = blockIdx.y * 64;
  const int b  = blockIdx.z;
  const BF* Bt = Y + (size_t)b * NTOK * CDIM;

  v8f acc[4][4] = {};
  for (int k0 = 0; k0 < CDIM; k0 += 32) {
    v16bf af[4], bfr[4];
#pragma unroll
    for (int mi = 0; mi < 4; ++mi)
      af[mi] = load_a_frag(Wbf, m0 + mi * 16 + ln, CDIM, k0, half);
#pragma unroll
    for (int ni = 0; ni < 4; ++ni)
      bfr[ni] = load_b_frag(Bt, n0 + ni * 16 + ln, CDIM, k0, half);
#pragma unroll
    for (int mi = 0; mi < 4; ++mi)
#pragma unroll
      for (int ni = 0; ni < 4; ++ni)
        acc[mi][ni] = wmma_bf16(af[mi], bfr[ni], acc[mi][ni]);
  }

#pragma unroll
  for (int mi = 0; mi < 4; ++mi)
#pragma unroll
    for (int ni = 0; ni < 4; ++ni)
#pragma unroll
      for (int j = 0; j < 8; ++j) {
        int m = m0 + mi * 16 + half * 8 + j;
        int n = n0 + ni * 16 + ln;
        out[((size_t)b * CDIM + m) * NTOK + n] = acc[mi][ni][j] + bout[m];
      }
}

// ---------------------------------------------------------------------------
// Launch
// ---------------------------------------------------------------------------
extern "C" void kernel_launch(void* const* d_in, const int* in_sizes, int n_in,
                              void* d_out, int out_size, void* d_ws,
                              size_t ws_size, hipStream_t stream) {
  const float* fmap  = (const float*)d_in[0];
  const float* Wqkvl = (const float*)d_in[1];
  const float* Wout  = (const float*)d_in[2];
  const float* bout  = (const float*)d_in[3];
  float*       out   = (float*)d_out;

  char*  ws  = (char*)d_ws;
  size_t off = 0;
  auto alloc = [&](size_t bytes) -> void* {
    void* p = ws + off;
    off += (bytes + 255) & ~(size_t)255;
    return p;
  };
  BF* Wqbf = (BF*)alloc((size_t)4 * CDIM * CDIM * 2);          // 2 MB
  BF* Wobf = (BF*)alloc((size_t)CDIM * CDIM * 2);              // 0.5 MB
  BF* fT   = (BF*)alloc((size_t)BATCH * NTOK * CDIM * 2);      // 8 MB
  BF* qb   = (BF*)alloc((size_t)BATCH * HEADS * NTOK * DHEAD * 2);
  BF* kb   = (BF*)alloc((size_t)BATCH * HEADS * NTOK * DHEAD * 2);
  BF* vT   = (BF*)alloc((size_t)BATCH * HEADS * NTOK * DHEAD * 2);
  BF* Y    = (BF*)alloc((size_t)BATCH * NTOK * CDIM * 2);      // 8 MB

  int nW = 4 * CDIM * CDIM;
  cvt_kernel<<<(nW + 255) / 256, 256, 0, stream>>>(Wqkvl, Wqbf, nW);
  int nWo = CDIM * CDIM;
  cvt_kernel<<<(nWo + 255) / 256, 256, 0, stream>>>(Wout, Wobf, nWo);
  int nF = BATCH * NTOK * CDIM;
  transpose_fmap_kernel<<<(nF + 255) / 256, 256, 0, stream>>>(fmap, fT);

  qkvl_kernel<<<dim3(NTOK / 64, (4 * CDIM) / 64, BATCH), 32, 0, stream>>>(
      Wqbf, fT, qb, kb, vT, Y);

  attn_kernel<<<dim3(BATCH * HEADS, NTOK / 64), 128, 0, stream>>>(qb, kb, vT,
                                                                  Y);

  outproj_kernel<<<dim3(NTOK / 64, CDIM / 64, BATCH), 32, 0, stream>>>(
      Wobf, Y, bout, out);
}
